// RandmanSNN_74474732912641
// MI455X (gfx1250) — matrix operand order repaired and checked by reading, not compiled, not run on hardware
//
#include <hip/hip_runtime.h>
#include <hip/hip_bf16.h>

// ---------------------------------------------------------------------------
// SNN forward on MI455X (gfx1250): bf16 WMMA GEMMs + fused LIF epilogue.
// Per timestep: layer1 kernel (x_t@W1^T + spk1@W1r^T -> mem1/spk1),
//               layer2 kernel (spk1@W2^T + spk2@W2r^T -> mem2/spk2),
//               readout kernel (mem_out = b*mem_out + spk2@Wout^T).
// Spike buffers ping-pong across t to avoid intra-kernel RAW races.
// Weights (~7MB bf16) stay resident in the 192MB L2 across all 100 steps;
// the 105MB spike record is written with nontemporal stores to keep it that
// way.
// ---------------------------------------------------------------------------

typedef __bf16 bf16;
typedef __attribute__((ext_vector_type(8)))  bf16  v8bf;
typedef __attribute__((ext_vector_type(16))) bf16  v16bf;
typedef __attribute__((ext_vector_type(8)))  float v8f;

#define BATCH 256
#define TSTEPS 100
#define NIN 512
#define HID 1024   // H1 == H2 == 1024
#define NOUT 10

#define NT 2       // N-tiles (16 cols each) per wave: 512 waves / 64 blocks

// ---------------------------------------------------------------------------
// fp32 -> bf16 conversion (weights + input sequence)
// ---------------------------------------------------------------------------
__global__ __launch_bounds__(256) void cvt_f32_bf16_kernel(
    const float* __restrict__ in, bf16* __restrict__ out, int n) {
  int i = blockIdx.x * blockDim.x + threadIdx.x;
  if (i < n) out[i] = (bf16)in[i];
}

// ---------------------------------------------------------------------------
// One wave accumulates a 16x(NT*16) strip; NT accumulators share one A frag.
// A: [M x K] with row stride lda (x is time-strided: lda = T*NI).
// W: [N x K] row-major; B = W^T so a B fragment is 16 contiguous bf16 from a
// row of W.
// A layout:            lane<16 -> M=lane,   K {k..k+15};
//                      lane>=16 -> M=lane-16, K {k+16..k+31}.
// B layout:            lane<16 -> N=n, K {k..k+15}; lane>=16 -> K {k+16..k+31}
// ---------------------------------------------------------------------------
__device__ __forceinline__ void gemm_acc(const bf16* __restrict__ A, int lda,
                                         const bf16* __restrict__ W, int ldw,
                                         int K, int m0, int n0, int lane,
                                         v8f acc[NT]) {
  const int koA = (lane & 16) ? 16 : 0;
  const int koB = (lane & 16) ? 16 : 0;
  const bf16* arow = A + (size_t)(m0 + (lane & 15)) * lda + koA;
  for (int k = 0; k < K; k += 32) {
    v16bf a = *(const v16bf*)(arow + k);
#pragma unroll
    for (int j = 0; j < NT; ++j) {
      const bf16* bp =
          W + (size_t)(n0 + j * 16 + (lane & 15)) * ldw + k + koB;
      v16bf b = *(const v16bf*)bp;
      acc[j] = __builtin_amdgcn_wmma_f32_16x16x32_bf16(
          false, a, false, b, (short)0, acc[j], false, false);
    }
  }
}

// ---------------------------------------------------------------------------
// Fused SNN layer: cur = A1@Wff^T + Aprev@Wrec^T; LIF update + spike.
// Grid: 64 blocks x 256 threads = 512 waves = 16 Mtiles x 32 Ngroups.
// ---------------------------------------------------------------------------
__global__ __launch_bounds__(256) void snn_layer_kernel(
    const bf16* __restrict__ A1, int lda1, int K1,  // input [BATCH x K1], stride lda1
    const bf16* __restrict__ Wff,                   // [HID x K1]
    const bf16* __restrict__ Aprev,                 // prev spikes [BATCH x HID]
    const bf16* __restrict__ Wrec,                  // [HID x HID]
    const float* __restrict__ beta_p,
    float* __restrict__ mem,                        // [BATCH x HID] fp32 state
    bf16* __restrict__ spk_out,                     // [BATCH x HID] next-step in
    float* __restrict__ spk_out_f32)                // [BATCH x HID] or nullptr
{
  const int lane = threadIdx.x & 31;
  const int wg = (blockIdx.x * blockDim.x + threadIdx.x) >> 5;
  const int m0 = (wg >> 5) << 4;          // 16 M tiles
  const int n0 = (wg & 31) * (NT * 16);   // 32 N groups of NT*16 cols

  float beta = beta_p[0];
  beta = fminf(fmaxf(beta, 0.0f), 1.0f);

  v8f acc[NT] = {};
  gemm_acc(A1, lda1, Wff, K1, K1, m0, n0, lane, acc);      // feedforward
  gemm_acc(Aprev, HID, Wrec, HID, HID, m0, n0, lane, acc); // recurrent

  // Epilogue: C/D layout -> lane<16: rows m0..m0+7, lane>=16: rows m0+8..15.
  const int rbase = m0 + ((lane & 16) ? 8 : 0);
#pragma unroll
  for (int j = 0; j < NT; ++j) {
    const int col = n0 + j * 16 + (lane & 15);
#pragma unroll
    for (int r = 0; r < 8; ++r) {
      const size_t idx = (size_t)(rbase + r) * HID + col;
      float m_old = mem[idx];
      float reset = (m_old > 1.0f) ? 1.0f : 0.0f;            // THR = 1
      float m_new = beta * m_old + acc[j][r] - reset;
      mem[idx] = m_new;
      float s = (m_new > 1.0f) ? 1.0f : 0.0f;                // spike(m-THR)
      spk_out[idx] = (bf16)s;
      if (spk_out_f32)
        __builtin_nontemporal_store(s, spk_out_f32 + idx);   // stream, skip L2
    }
  }
}

// ---------------------------------------------------------------------------
// Readout: mem_out = clip(beta)*mem_out + spk2 @ Wout^T  (NO=10, tiny)
// ---------------------------------------------------------------------------
__global__ __launch_bounds__(256) void snn_readout_kernel(
    const bf16* __restrict__ spk2,   // [BATCH x HID]
    const bf16* __restrict__ Wout,   // [NOUT x HID]
    const float* __restrict__ beta_p,
    float* __restrict__ mem_out,     // [BATCH x NOUT]
    float* __restrict__ out)         // d_out slice [BATCH x NOUT]
{
  int i = blockIdx.x * blockDim.x + threadIdx.x;
  if (i >= BATCH * NOUT) return;
  int b = i / NOUT, o = i % NOUT;
  float beta = fminf(fmaxf(beta_p[0], 0.0f), 1.0f);
  const bf16* s = spk2 + (size_t)b * HID;
  const bf16* w = Wout + (size_t)o * HID;
  float acc = 0.0f;
  for (int k = 0; k < HID; k += 8) {
    v8bf sv = *(const v8bf*)(s + k);
    v8bf wv = *(const v8bf*)(w + k);
#pragma unroll
    for (int q = 0; q < 8; ++q) acc += (float)sv[q] * (float)wv[q];
  }
  float m = beta * mem_out[i] + acc;
  mem_out[i] = m;
  __builtin_nontemporal_store(m, out + i);
}

// ---------------------------------------------------------------------------
extern "C" void kernel_launch(void* const* d_in, const int* in_sizes, int n_in,
                              void* d_out, int out_size, void* d_ws,
                              size_t ws_size, hipStream_t stream) {
  const float* x     = (const float*)d_in[0];   // [B,T,NI]
  const float* W1    = (const float*)d_in[1];   // [H1,NI]
  const float* W1r   = (const float*)d_in[2];   // [H1,H1]
  const float* W2    = (const float*)d_in[3];   // [H2,H1]
  const float* W2r   = (const float*)d_in[4];   // [H2,H2]
  const float* Wout  = (const float*)d_in[5];   // [NO,H2]
  const float* beta1 = (const float*)d_in[6];
  const float* beta2 = (const float*)d_in[7];
  const float* betao = (const float*)d_in[8];

  // ---- workspace carve (256B aligned); total ~38 MB ----
  char* p = (char*)d_ws;
  auto carve = [&](size_t bytes) {
    char* r = p;
    p += (bytes + 255) & ~(size_t)255;
    return r;
  };
  bf16* xb    = (bf16*)carve((size_t)BATCH * TSTEPS * NIN * sizeof(bf16));
  bf16* W1b   = (bf16*)carve((size_t)HID * NIN * sizeof(bf16));
  bf16* W1rb  = (bf16*)carve((size_t)HID * HID * sizeof(bf16));
  bf16* W2b   = (bf16*)carve((size_t)HID * HID * sizeof(bf16));
  bf16* W2rb  = (bf16*)carve((size_t)HID * HID * sizeof(bf16));
  bf16* Woutb = (bf16*)carve((size_t)NOUT * HID * sizeof(bf16));
  float* mem1 = (float*)carve((size_t)BATCH * HID * sizeof(float));
  float* mem2 = (float*)carve((size_t)BATCH * HID * sizeof(float));
  float* memo = (float*)carve((size_t)BATCH * NOUT * sizeof(float));
  bf16* spk1buf[2] = {
      (bf16*)carve((size_t)BATCH * HID * sizeof(bf16)),
      (bf16*)carve((size_t)BATCH * HID * sizeof(bf16))};
  bf16* spk2buf[2] = {
      (bf16*)carve((size_t)BATCH * HID * sizeof(bf16)),
      (bf16*)carve((size_t)BATCH * HID * sizeof(bf16))};

  // ---- zero state (memset nodes are graph-capture safe) ----
  hipMemsetAsync(mem1, 0, (size_t)BATCH * HID * sizeof(float), stream);
  hipMemsetAsync(mem2, 0, (size_t)BATCH * HID * sizeof(float), stream);
  hipMemsetAsync(memo, 0, (size_t)BATCH * NOUT * sizeof(float), stream);
  hipMemsetAsync(spk1buf[0], 0, (size_t)BATCH * HID * sizeof(bf16), stream);
  hipMemsetAsync(spk1buf[1], 0, (size_t)BATCH * HID * sizeof(bf16), stream);
  hipMemsetAsync(spk2buf[0], 0, (size_t)BATCH * HID * sizeof(bf16), stream);
  hipMemsetAsync(spk2buf[1], 0, (size_t)BATCH * HID * sizeof(bf16), stream);

  // ---- fp32 -> bf16 conversions (redone each call for determinism) ----
  auto cvt = [&](const float* src, bf16* dst, size_t n) {
    cvt_f32_bf16_kernel<<<(unsigned)((n + 255) / 256), 256, 0, stream>>>(
        src, dst, (int)n);
  };
  cvt(x,    xb,    (size_t)BATCH * TSTEPS * NIN);
  cvt(W1,   W1b,   (size_t)HID * NIN);
  cvt(W1r,  W1rb,  (size_t)HID * HID);
  cvt(W2,   W2b,   (size_t)HID * HID);
  cvt(W2r,  W2rb,  (size_t)HID * HID);
  cvt(Wout, Woutb, (size_t)NOUT * HID);

  float* spk1_rec = (float*)d_out;                            // [T,B,H1]
  float* memo_rec = spk1_rec + (size_t)TSTEPS * BATCH * HID;  // [T,B,NO]

  // 512 waves = 16 Mtiles x 32 Ngroups -> 64 blocks of 8 waves
  const dim3 lgrid(64), lblk(256);
  const dim3 rgrid((BATCH * NOUT + 255) / 256), rblk(256);

  for (int t = 0; t < TSTEPS; ++t) {
    const int cur = t & 1, nxt = (t + 1) & 1;
    // x layout [B,T,NI]: row for batch b at time t is (b*TSTEPS + t)*NIN.
    // -> base xb + t*NIN, row stride lda = TSTEPS*NIN.
    const bf16* xt = xb + (size_t)t * NIN;

    // Layer 1
    snn_layer_kernel<<<lgrid, lblk, 0, stream>>>(
        xt, TSTEPS * NIN, NIN, W1b, spk1buf[cur], W1rb, beta1, mem1,
        spk1buf[nxt], spk1_rec + (size_t)t * BATCH * HID);

    // Layer 2
    snn_layer_kernel<<<lgrid, lblk, 0, stream>>>(
        spk1buf[nxt], HID, HID, W2b, spk2buf[cur], W2rb, beta2, mem2,
        spk2buf[nxt], nullptr);

    // Readout
    snn_readout_kernel<<<rgrid, rblk, 0, stream>>>(
        spk2buf[nxt], Woutb, betao, memo, memo_rec + (size_t)t * BATCH * NOUT);
  }
  (void)in_sizes; (void)n_in; (void)out_size; (void)ws_size;
}
